// All2AllCostVolume_75196287418836
// MI455X (gfx1250) — compile-verified, matrix operand order checked by reading, not run on hardware
//
#include <hip/hip_runtime.h>
#include <hip/hip_bf16.h>
#include <math.h>

// ---------------------------------------------------------------------------
// Problem constants (match reference)
// ---------------------------------------------------------------------------
constexpr int BATCH = 2;
constexpr int NPTS  = 4096;   // N
constexpr int MPTS  = 4096;   // M
constexpr int CCH   = 64;     // C
constexpr int KQn   = 16;
constexpr int KSn   = 8;

constexpr int R1 = BATCH * NPTS * KQn;   // 131072 rows, stage 1
constexpr int R2 = BATCH * NPTS * KSn;   // 65536 rows, stage 2

constexpr int LDH0 = 204;     // H0 row stride (202 cols padded to x4 for b128 loads)

typedef __attribute__((ext_vector_type(16))) _Float16 v16h;
typedef __attribute__((ext_vector_type(8)))  float    v8f;

static inline int ceil_div(int a, int b) { return (a + b - 1) / b; }

// ---------------------------------------------------------------------------
// Weight pre-pack: rearrange W[cin,cout] (f32) into per-lane WMMA B-fragment
// order as f16, zero-padded to K multiple of 32.
// Layout: Wp[((kk*nCT + ct)*32 + lane)*16 + e], where
//   k = kk*32 + e + ((e&8)?8:0) + 8*(lane>>4), col = ct*16 + (lane&15).
// ---------------------------------------------------------------------------
__global__ void pack_w_kernel(const float* __restrict__ W, int cin, int cout,
                              _Float16* __restrict__ Wp)
{
    const int nCT   = cout >> 4;
    const int nK    = (cin + 31) >> 5;
    const int total = nK * nCT * 512;
    const int t = blockIdx.x * blockDim.x + threadIdx.x;
    if (t >= total) return;
    const int e    = t & 15;
    const int lane = (t >> 4) & 31;
    const int kt   = t >> 9;            // kk*nCT + ct
    const int ct   = kt % nCT;
    const int kk   = kt / nCT;
    const int k    = kk * 32 + e + ((e & 8) ? 8 : 0) + ((lane >> 4) * 8);
    const int col  = ct * 16 + (lane & 15);
    const float v  = (k < cin) ? W[(size_t)k * cout + col] : 0.0f;
    Wp[t] = (_Float16)v;
}

// ---------------------------------------------------------------------------
// Fused GEMM (Y = act(X)·W + bias) + deterministic BatchNorm partial stats.
// Compile-time CIN/COUT/HASAFF: fully unrolled, A fragments register-resident
// across all column tiles, B fragments are single 32B packed loads.
// act(x) = HASAFF ? relu(x*inAff[c] + inAff[CIN+c]) : x   (folded prev BN+ReLU)
// ---------------------------------------------------------------------------
template <int CIN, int COUT, bool HASAFF>
__global__ void gemm_bn_kernel(const float* __restrict__ X, int ldx,
                               const _Float16* __restrict__ Wp,
                               const float* __restrict__ bias,
                               const float* __restrict__ inAff,
                               float* __restrict__ Y,       // [R, COUT]
                               float* __restrict__ stats,   // [nBlocks, 2*COUT]
                               int R)
{
    constexpr int KFULL = CIN >> 5;
    constexpr int TAIL  = CIN & 31;
    constexpr int NFRAG = KFULL + (TAIL ? 1 : 0);
    constexpr int NCT   = COUT >> 4;

    __shared__ float sstat[4 * 2 * COUT];
    const int tid  = threadIdx.x;
    const int lane = tid & 31;           // wave32
    const int wave = tid >> 5;
    for (int j = tid; j < 4 * 2 * COUT; j += 128) sstat[j] = 0.0f;
    __syncthreads();

    const int rowBase = blockIdx.x * 64 + wave * 16;
    if (rowBase < R) {
        const int rA = rowBase + (lane & 15);
        const int kg = (lane >> 4) * 8;
        const float* __restrict__ xrow = X + (size_t)rA * ldx;

        // ---- load all A fragments once (register resident) ----
        v16h afrag[NFRAG];
#pragma unroll
        for (int kk = 0; kk < KFULL; ++kk) {
            const int base = kk * 32 + kg;   // 16B-aligned (ldx % 4 == 0)
            const float4 x0 = *reinterpret_cast<const float4*>(xrow + base);
            const float4 x1 = *reinterpret_cast<const float4*>(xrow + base + 4);
            const float4 x2 = *reinterpret_cast<const float4*>(xrow + base + 16);
            const float4 x3 = *reinterpret_cast<const float4*>(xrow + base + 20);
            float xv[16] = { x0.x, x0.y, x0.z, x0.w,  x1.x, x1.y, x1.z, x1.w,
                             x2.x, x2.y, x2.z, x2.w,  x3.x, x3.y, x3.z, x3.w };
            if constexpr (HASAFF) {
                const float4 a0 = *reinterpret_cast<const float4*>(inAff + base);
                const float4 a1 = *reinterpret_cast<const float4*>(inAff + base + 4);
                const float4 a2 = *reinterpret_cast<const float4*>(inAff + base + 16);
                const float4 a3 = *reinterpret_cast<const float4*>(inAff + base + 20);
                const float4 b0 = *reinterpret_cast<const float4*>(inAff + CIN + base);
                const float4 b1 = *reinterpret_cast<const float4*>(inAff + CIN + base + 4);
                const float4 b2 = *reinterpret_cast<const float4*>(inAff + CIN + base + 16);
                const float4 b3 = *reinterpret_cast<const float4*>(inAff + CIN + base + 20);
                const float av[16] = { a0.x,a0.y,a0.z,a0.w, a1.x,a1.y,a1.z,a1.w,
                                       a2.x,a2.y,a2.z,a2.w, a3.x,a3.y,a3.z,a3.w };
                const float bv[16] = { b0.x,b0.y,b0.z,b0.w, b1.x,b1.y,b1.z,b1.w,
                                       b2.x,b2.y,b2.z,b2.w, b3.x,b3.y,b3.z,b3.w };
#pragma unroll
                for (int i = 0; i < 16; ++i)
                    xv[i] = fmaxf(0.0f, fmaf(xv[i], av[i], bv[i]));
            }
            v16h a;
#pragma unroll
            for (int i = 0; i < 16; ++i) a[i] = (_Float16)xv[i];
            afrag[kk] = a;
        }
        if constexpr (TAIL != 0) {
            v16h a{};
#pragma unroll
            for (int e = 0; e < 16; ++e) {
                const int k = KFULL * 32 + e + ((e & 8) ? 8 : 0) + kg;
                float x = 0.0f;
                if (k < CIN) {
                    x = xrow[k];
                    if constexpr (HASAFF)
                        x = fmaxf(0.0f, fmaf(x, inAff[k], inAff[CIN + k]));
                }
                a[e] = (_Float16)x;
            }
            afrag[KFULL] = a;
        }

        // ---- column tiles: packed-B loads + WMMA ----
#pragma unroll
        for (int ct = 0; ct < NCT; ++ct) {
            v8f acc = {};
#pragma unroll
            for (int kk = 0; kk < NFRAG; ++kk) {
                const v16h b = *reinterpret_cast<const v16h*>(
                    Wp + (((size_t)kk * NCT + ct) << 9) + (lane << 4));
                acc = __builtin_amdgcn_wmma_f32_16x16x32_f16(
                        false, afrag[kk], false, b, (short)0, acc, false, false);
            }
            const int   ch  = ct * 16 + (lane & 15);
            const float bs  = bias[ch];
            const int   rD0 = rowBase + ((lane >> 4) << 3);
            float s = 0.0f, q = 0.0f;
#pragma unroll
            for (int r = 0; r < 8; ++r) {
                const float y = acc[r] + bs;
                Y[(size_t)(rD0 + r) * COUT + ch] = y;
                s += y; q += y * y;
            }
            s += __shfl_xor(s, 16, 32);
            q += __shfl_xor(q, 16, 32);
            if (lane < 16) {
                sstat[wave * 2 * COUT + 2 * ch]     = s;
                sstat[wave * 2 * COUT + 2 * ch + 1] = q;
            }
        }
    }
    __syncthreads();
    for (int j = tid; j < 2 * COUT; j += 128) {
        const float a = sstat[j] + sstat[2 * COUT + j] +
                        sstat[4 * COUT + j] + sstat[6 * COUT + j];
        stats[(size_t)blockIdx.x * (2 * COUT) + j] = a;
    }
}

// Parallel block-partial reduce -> per-channel affine (A,B): one block/channel.
__global__ void bn_finalize_kernel(const float* __restrict__ stats, int nBlocks, int cout,
                                   float invCount,
                                   const float* __restrict__ gg,
                                   const float* __restrict__ be,
                                   float* __restrict__ aff)   // [2*cout]
{
    __shared__ float rs[256], rq[256];
    const int c = blockIdx.x;
    float s = 0.0f, q = 0.0f;
    for (int b = threadIdx.x; b < nBlocks; b += 256) {
        s += stats[(size_t)b * (2 * cout) + 2 * c];
        q += stats[(size_t)b * (2 * cout) + 2 * c + 1];
    }
    rs[threadIdx.x] = s; rq[threadIdx.x] = q;
    __syncthreads();
    for (int st = 128; st > 0; st >>= 1) {
        if (threadIdx.x < st) {
            rs[threadIdx.x] += rs[threadIdx.x + st];
            rq[threadIdx.x] += rq[threadIdx.x + st];
        }
        __syncthreads();
    }
    if (threadIdx.x == 0) {
        const float m = rs[0] * invCount;
        const float v = fmaxf(rq[0] * invCount - m * m, 0.0f);
        const float A = gg[c] * rsqrtf(v + 1e-5f);
        aff[c]        = A;
        aff[cout + c] = be[c] - m * A;
    }
}

// ---------------------------------------------------------------------------
// KNN: LDS-tiled candidates, per-thread top-K insertion registers.
// Requires: (BATCH*nq) % 128 == 0, np % 128 == 0, nq % 128 == 0 (true here).
// ---------------------------------------------------------------------------
template <int K>
__global__ void knn_kernel(const float* __restrict__ qxyz,  // [B, nq, 3] queries
                           const float* __restrict__ pxyz,  // [B, np, 3] database
                           int nq, int np, int* __restrict__ outIdx)
{
    __shared__ float sx[128], sy[128], sz[128];
    const int t = blockIdx.x * 128 + threadIdx.x;   // all threads in block: same b
    const int b = t / nq;
    const float qx = qxyz[(size_t)t * 3 + 0];
    const float qy = qxyz[(size_t)t * 3 + 1];
    const float qz = qxyz[(size_t)t * 3 + 2];

    float bd[K]; int bi[K];
#pragma unroll
    for (int i = 0; i < K; ++i) { bd[i] = 3.0e38f; bi[i] = 0; }

    const float* __restrict__ pb = pxyz + (size_t)b * np * 3;
    for (int m0 = 0; m0 < np; m0 += 128) {
        __syncthreads();
        const int mi = m0 + threadIdx.x;
        sx[threadIdx.x] = pb[3 * mi + 0];
        sy[threadIdx.x] = pb[3 * mi + 1];
        sz[threadIdx.x] = pb[3 * mi + 2];
        __syncthreads();
        for (int j = 0; j < 128; ++j) {
            const float dx = sx[j] - qx;
            const float dy = sy[j] - qy;
            const float dz = sz[j] - qz;
            const float d  = dx * dx + dy * dy + dz * dz;
            if (d < bd[K - 1]) {
                bd[K - 1] = d; bi[K - 1] = m0 + j;
#pragma unroll
                for (int i = K - 1; i > 0; --i) {
                    if (bd[i] < bd[i - 1]) {
                        float td = bd[i]; bd[i] = bd[i - 1]; bd[i - 1] = td;
                        int   ti = bi[i]; bi[i] = bi[i - 1]; bi[i - 1] = ti;
                    }
                }
            }
        }
    }
#pragma unroll
    for (int i = 0; i < K; ++i) outIdx[(size_t)t * K + i] = bi[i];
}

// ---------------------------------------------------------------------------
// Stage-1 feature build (H0 cols 0..137): xyz_cat(10) | pi_n(64) | qi_n(64)
// H0 row stride = LDH0 (=204).
// ---------------------------------------------------------------------------
__global__ void build_h0_kernel(const float* __restrict__ wxyz,
                                const float* __restrict__ wpts,
                                const float* __restrict__ f2xyz,
                                const float* __restrict__ f2pts,
                                const int* __restrict__ idxq,
                                float* __restrict__ H0)
{
    const int t = blockIdx.x * blockDim.x + threadIdx.x;
    if (t >= R1) return;
    const int bn  = t / KQn;
    const int b   = bn / NPTS;
    const int idx = idxq[t];

    const float* p  = wxyz + (size_t)bn * 3;
    const float* qp = f2xyz + ((size_t)b * MPTS + idx) * 3;
    const float d0 = qp[0] - p[0], d1 = qp[1] - p[1], d2 = qp[2] - p[2];
    const float euc = sqrtf(d0 * d0 + d1 * d1 + d2 * d2 + 1e-20f);

    float* o = H0 + (size_t)t * LDH0;
    o[0] = p[0]; o[1] = p[1]; o[2] = p[2];
    o[3] = qp[0]; o[4] = qp[1]; o[5] = qp[2];
    o[6] = d0; o[7] = d1; o[8] = d2; o[9] = euc;

    {   // norm_feat(warped_points[b,n])  (mean, std ddof=1 over C=64)
        const float* pv = wpts + (size_t)bn * CCH;
        float s = 0.0f, q = 0.0f;
        for (int c = 0; c < CCH; ++c) { const float x = pv[c]; s += x; q += x * x; }
        const float m   = s * (1.0f / CCH);
        const float var = (q - CCH * m * m) * (1.0f / (CCH - 1));
        const float inv = rsqrtf(fmaxf(var, 1e-30f));
        for (int c = 0; c < CCH; ++c) o[10 + c] = (pv[c] - m) * inv;
    }
    {   // norm_feat(f2_points[b, idx])
        const float* qv = f2pts + ((size_t)b * MPTS + idx) * CCH;
        float s = 0.0f, q = 0.0f;
        for (int c = 0; c < CCH; ++c) { const float x = qv[c]; s += x; q += x * x; }
        const float m   = s * (1.0f / CCH);
        const float var = (q - CCH * m * m) * (1.0f / (CCH - 1));
        const float inv = rsqrtf(fmaxf(var, 1e-30f));
        for (int c = 0; c < CCH; ++c) o[74 + c] = (qv[c] - m) * inv;
    }
}

// fd1max[b,k,c] = max over n of pi_n*qi_n  (read back from H0 slices)
__global__ void fd1max_kernel(const float* __restrict__ H0, float* __restrict__ out)
{
    __shared__ float red[128];
    const int bk = blockIdx.x;          // 0..B*KQ-1
    const int c  = blockIdx.y;          // 0..63
    const int b  = bk / KQn, k = bk % KQn;
    float mx = -3.0e38f;
    for (int n = threadIdx.x; n < NPTS; n += 128) {
        const size_t row = ((size_t)(b * NPTS + n)) * KQn + k;
        const float pi = H0[row * LDH0 + 10 + c];
        const float qi = H0[row * LDH0 + 74 + c];
        mx = fmaxf(mx, pi * qi);
    }
    red[threadIdx.x] = mx;
    __syncthreads();
    for (int s = 64; s > 0; s >>= 1) {
        if (threadIdx.x < s) red[threadIdx.x] = fmaxf(red[threadIdx.x], red[threadIdx.x + s]);
        __syncthreads();
    }
    if (threadIdx.x == 0) out[(size_t)bk * CCH + c] = red[0];
}

// H0 cols 138..201 = relu(affR( pi_rev(fd1max) )), broadcast over n
__global__ void fill_fd1_kernel(const float* __restrict__ Yrev,
                                const float* __restrict__ affR,
                                float* __restrict__ H0)
{
    const int t = blockIdx.x * blockDim.x + threadIdx.x;
    if (t >= R1 * CCH) return;
    const int c   = t & 63;
    const int row = t >> 6;
    const int k   = row % KQn;
    const int b   = row / (NPTS * KQn);
    const float y = Yrev[((size_t)(b * KQn + k)) * CCH + c];
    H0[(size_t)row * LDH0 + 138 + c] = fmaxf(0.0f, affR[c] * y + affR[CCH + c]);
}

// G0 = [ relu(affE(Yenc)) | relu(aff13(Y3)) ]   -> [R1, 128]
__global__ void concat_g0_kernel(const float* __restrict__ Yenc, const float* __restrict__ affE,
                                 const float* __restrict__ Y3,   const float* __restrict__ aff3,
                                 float* __restrict__ G0)
{
    const int t = blockIdx.x * blockDim.x + threadIdx.x;
    if (t >= R1 * CCH) return;
    const int c = t & 63;
    const int r = t >> 6;
    G0[(size_t)r * 128 + c]      = fmaxf(0.0f, affE[c] * Yenc[(size_t)r * CCH + c] + affE[CCH + c]);
    G0[(size_t)r * 128 + 64 + c] = fmaxf(0.0f, aff3[c] * Y3[(size_t)r * CCH + c] + aff3[CCH + c]);
}

// Stage-1 softmax over KQ + weighted sum:  feat1 = sum_k softmax(g)*h
__global__ void smws1_kernel(const float* __restrict__ Z2, const float* __restrict__ affZ,
                             const float* __restrict__ Y3, const float* __restrict__ aff3,
                             float* __restrict__ feat1)
{
    const int t = blockIdx.x * blockDim.x + threadIdx.x;
    if (t >= BATCH * NPTS * CCH) return;
    const int c  = t & 63;
    const int bn = t >> 6;
    float gv[KQn], hv[KQn], mx = -3.0e38f;
#pragma unroll
    for (int k = 0; k < KQn; ++k) {
        const size_t row = (size_t)bn * KQn + k;
        const float g = fmaxf(0.0f, affZ[c] * Z2[row * CCH + c] + affZ[CCH + c]);
        const float h = fmaxf(0.0f, aff3[c] * Y3[row * CCH + c] + aff3[CCH + c]);
        gv[k] = g; hv[k] = h; mx = fmaxf(mx, g);
    }
    float den = 0.0f, num = 0.0f;
#pragma unroll
    for (int k = 0; k < KQn; ++k) {
        const float e = expf(gv[k] - mx);
        den += e; num += e * hv[k];
    }
    feat1[t] = num / den;
}

// Stage-2 xyz features: [pc_xyz_n | pc_xyz_g | d | euc]  -> [R2, 10]
__global__ void build_xyz2_kernel(const float* __restrict__ wxyz,
                                  const int* __restrict__ idxs,
                                  float* __restrict__ X2)
{
    const int t = blockIdx.x * blockDim.x + threadIdx.x;
    if (t >= R2) return;
    const int bn  = t / KSn;
    const int b   = bn / NPTS;
    const int idx = idxs[t];
    const float* p = wxyz + (size_t)bn * 3;
    const float* g = wxyz + ((size_t)b * NPTS + idx) * 3;
    const float d0 = g[0] - p[0], d1 = g[1] - p[1], d2 = g[2] - p[2];
    float* o = X2 + (size_t)t * 10;
    o[0] = p[0]; o[1] = p[1]; o[2] = p[2];
    o[3] = g[0]; o[4] = g[1]; o[5] = g[2];
    o[6] = d0; o[7] = d1; o[8] = d2;
    o[9] = sqrtf(d0 * d0 + d1 * d1 + d2 * d2 + 1e-20f);
}

// F2 = [ relu(affE2(Yenc2)) | warped_points | group(feat1, idx_s) ] -> [R2, 192]
__global__ void concat_f2_kernel(const float* __restrict__ Yenc2, const float* __restrict__ affE2,
                                 const float* __restrict__ wpts,
                                 const float* __restrict__ feat1,
                                 const int* __restrict__ idxs,
                                 float* __restrict__ F2)
{
    const int t = blockIdx.x * blockDim.x + threadIdx.x;
    if (t >= R2 * CCH) return;
    const int c  = t & 63;
    const int r2 = t >> 6;
    const int bn = r2 / KSn;
    const int b  = bn / NPTS;
    const int idx = idxs[r2];
    F2[(size_t)r2 * 192 + c]       = fmaxf(0.0f, affE2[c] * Yenc2[(size_t)r2 * CCH + c] + affE2[CCH + c]);
    F2[(size_t)r2 * 192 + 64 + c]  = wpts[(size_t)bn * CCH + c];
    F2[(size_t)r2 * 192 + 128 + c] = feat1[((size_t)b * NPTS + idx) * CCH + c];
}

// Final: out = sum_k softmax_k(relu(affU(U2))) * group(feat1, idx_s)
__global__ void final_kernel(const float* __restrict__ U2, const float* __restrict__ affU,
                             const float* __restrict__ feat1,
                             const int* __restrict__ idxs,
                             float* __restrict__ outp)
{
    const int t = blockIdx.x * blockDim.x + threadIdx.x;
    if (t >= BATCH * NPTS * CCH) return;
    const int c  = t & 63;
    const int bn = t >> 6;
    const int b  = bn / NPTS;
    float gv[KSn], hv[KSn], mx = -3.0e38f;
#pragma unroll
    for (int k = 0; k < KSn; ++k) {
        const size_t r2 = (size_t)bn * KSn + k;
        const float g = fmaxf(0.0f, affU[c] * U2[r2 * CCH + c] + affU[CCH + c]);
        const int idx = idxs[r2];
        gv[k] = g;
        hv[k] = feat1[((size_t)b * NPTS + idx) * CCH + c];
        mx = fmaxf(mx, g);
    }
    float den = 0.0f, num = 0.0f;
#pragma unroll
    for (int k = 0; k < KSn; ++k) {
        const float e = expf(gv[k] - mx);
        den += e; num += e * hv[k];
    }
    outp[t] = num / den;
}

// ---------------------------------------------------------------------------
// Host launcher
// ---------------------------------------------------------------------------
template <int CIN, int COUT, bool HASAFF>
static void run_conv(const float* X, int ldx, const float* W, const float* bias,
                     const float* inAff, _Float16* Wp, float* Y, float* stats, int R,
                     const float* g, const float* be, float* aff, hipStream_t stream)
{
    const int nK    = (CIN + 31) / 32;
    const int packN = nK * (COUT / 16) * 512;
    pack_w_kernel<<<ceil_div(packN, 256), 256, 0, stream>>>(W, CIN, COUT, Wp);
    const int nBlocks = ceil_div(R, 64);
    gemm_bn_kernel<CIN, COUT, HASAFF><<<nBlocks, 128, 0, stream>>>(
        X, ldx, Wp, bias, inAff, Y, stats, R);
    bn_finalize_kernel<<<COUT, 256, 0, stream>>>(stats, nBlocks, COUT, 1.0f / (float)R, g, be, aff);
}

extern "C" void kernel_launch(void* const* d_in, const int* in_sizes, int n_in,
                              void* d_out, int out_size, void* d_ws, size_t ws_size,
                              hipStream_t stream)
{
    (void)in_sizes; (void)n_in; (void)out_size; (void)ws_size;

    // Inputs in setup_inputs() dict (insertion) order; params leaves in
    // insertion order: mlp1[0..2]{w,b,g,be}, pi_enc, mlp2[0..1], pc_enc,
    // mlp2_new[0..1], pi_rev.
    const float* wxyz  = (const float*)d_in[0];
    const float* wpts  = (const float*)d_in[1];
    const float* f2xyz = (const float*)d_in[2];
    const float* f2pts = (const float*)d_in[3];
    const float* P[40];
    for (int i = 0; i < 40; ++i) P[i] = (const float*)d_in[4 + i];
    // mlp1: P[0..11], pi_enc: P[12..15], mlp2: P[16..23],
    // pc_enc: P[24..27], mlp2_new: P[28..35], pi_rev: P[36..39]

    // ---- workspace arena (lifetime-based reuse) ----
    char* ws = (char*)d_ws;
    size_t off = 0;
    auto arena = [&](size_t bytes) -> char* {
        char* p = ws + off;
        off += (bytes + 255) & ~(size_t)255;
        return p;
    };
    int*      idxq  = (int*)     arena((size_t)R1 * 4);
    int*      idxs  = (int*)     arena((size_t)R2 * 4);
    float*    slotA = (float*)   arena((size_t)R1 * LDH0 * 4); // H0 -> G0 -> F2
    float*    slotB = (float*)   arena((size_t)R1 * 128 * 4);  // Y1 -> Z1
    float*    slotC = (float*)   arena((size_t)R1 * 64 * 4);   // Y2 -> U1
    float*    slotD = (float*)   arena((size_t)R1 * 64 * 4);   // Y3 (long-lived)
    float*    slotE = (float*)   arena((size_t)R1 * 64 * 4);   // Yenc -> {Yenc2, U2}
    float*    slotF = (float*)   arena((size_t)R1 * 64 * 4);   // Z2
    float*    feat1 = (float*)   arena((size_t)BATCH * NPTS * 64 * 4);
    float*    X2    = (float*)   arena((size_t)R2 * 10 * 4);
    float*    fd1m  = (float*)   arena(32 * 64 * 4);
    float*    Yrev  = (float*)   arena(32 * 64 * 4);
    float*    stats = (float*)   arena((size_t)2048 * 256 * 4);
    _Float16* Wpack = (_Float16*)arena(7 * 8 * 512 * 2);       // max packed weights
    float*    affR  = (float*)   arena(256 * 4);
    float*    aff11 = (float*)   arena(256 * 4);
    float*    aff12 = (float*)   arena(256 * 4);
    float*    aff13 = (float*)   arena(256 * 4);
    float*    affE  = (float*)   arena(256 * 4);
    float*    affZ1 = (float*)   arena(256 * 4);
    float*    affZ2 = (float*)   arena(256 * 4);
    float*    affE2 = (float*)   arena(256 * 4);
    float*    affU1 = (float*)   arena(256 * 4);
    float*    affU2 = (float*)   arena(256 * 4);

    float* H0    = slotA;
    float* G0    = slotA;                    // reuse after H0 dead
    float* F2    = slotA;                    // reuse after G0 dead
    float* Y1    = slotB, *Z1 = slotB;
    float* Y2    = slotC, *U1 = slotC;
    float* Y3    = slotD;
    float* Yenc  = slotE;
    float* Yenc2 = slotE;
    float* U2    = slotE + (size_t)R2 * 64;  // disjoint from Yenc2 within slotE
    float* Z2    = slotF;

    // ---- stage 1 ----
    knn_kernel<KQn><<<(BATCH * NPTS) / 128, 128, 0, stream>>>(wxyz, f2xyz, NPTS, MPTS, idxq);
    build_h0_kernel<<<ceil_div(R1, 128), 128, 0, stream>>>(wxyz, wpts, f2xyz, f2pts, idxq, H0);
    fd1max_kernel<<<dim3(BATCH * KQn, CCH), 128, 0, stream>>>(H0, fd1m);
    run_conv<64, 64, false>(fd1m, 64, P[36], P[37], nullptr, Wpack, Yrev, stats, 32,
                            P[38], P[39], affR, stream);
    fill_fd1_kernel<<<ceil_div(R1 * CCH, 256), 256, 0, stream>>>(Yrev, affR, H0);

    run_conv<202, 128, false>(H0, LDH0, P[0], P[1], nullptr, Wpack, Y1, stats, R1,
                              P[2], P[3], aff11, stream);
    run_conv<128, 64, true>(Y1, 128, P[4], P[5], aff11, Wpack, Y2, stats, R1,
                            P[6], P[7], aff12, stream);
    run_conv<64, 64, true>(Y2, 64, P[8], P[9], aff12, Wpack, Y3, stats, R1,
                           P[10], P[11], aff13, stream);
    run_conv<10, 64, false>(H0, LDH0, P[12], P[13], nullptr, Wpack, Yenc, stats, R1,
                            P[14], P[15], affE, stream);

    concat_g0_kernel<<<ceil_div(R1 * CCH, 256), 256, 0, stream>>>(Yenc, affE, Y3, aff13, G0);
    run_conv<128, 128, false>(G0, 128, P[16], P[17], nullptr, Wpack, Z1, stats, R1,
                              P[18], P[19], affZ1, stream);
    run_conv<128, 64, true>(Z1, 128, P[20], P[21], affZ1, Wpack, Z2, stats, R1,
                            P[22], P[23], affZ2, stream);
    smws1_kernel<<<ceil_div(BATCH * NPTS * CCH, 256), 256, 0, stream>>>(Z2, affZ2, Y3, aff13, feat1);

    // ---- stage 2 ----
    knn_kernel<KSn><<<(BATCH * NPTS) / 128, 128, 0, stream>>>(wxyz, wxyz, NPTS, NPTS, idxs);
    build_xyz2_kernel<<<ceil_div(R2, 128), 128, 0, stream>>>(wxyz, idxs, X2);
    run_conv<10, 64, false>(X2, 10, P[24], P[25], nullptr, Wpack, Yenc2, stats, R2,
                            P[26], P[27], affE2, stream);
    concat_f2_kernel<<<ceil_div(R2 * CCH, 256), 256, 0, stream>>>(Yenc2, affE2, wpts, feat1, idxs, F2);
    run_conv<192, 128, false>(F2, 192, P[28], P[29], nullptr, Wpack, U1, stats, R2,
                              P[30], P[31], affU1, stream);
    run_conv<128, 64, true>(U1, 128, P[32], P[33], affU1, Wpack, U2, stats, R2,
                            P[34], P[35], affU2, stream);
    final_kernel<<<ceil_div(BATCH * NPTS * CCH, 256), 256, 0, stream>>>(U2, affU2, feat1, idxs, (float*)d_out);
}
